// GroupQueryAttention_9216999817493
// MI455X (gfx1250) — matrix operand order repaired and checked by reading, not compiled, not run on hardware
//
#include <hip/hip_runtime.h>

// ---------------------------------------------------------------------------
// GQA fused pipeline for MI455X (gfx1250, wave32, WMMA bf16).
// B=2, S=2048, D=2048, NH=32, NKV=8, HD=64, G=4.
// ---------------------------------------------------------------------------

typedef __attribute__((ext_vector_type(16))) __bf16        v16bf;
typedef __attribute__((ext_vector_type(8)))  float         v8f;
typedef __attribute__((ext_vector_type(8)))  unsigned int  v8u;

static_assert(sizeof(v16bf) == 32 && sizeof(v8u) == 32, "frag size");

constexpr int B_   = 2;
constexpr int S_   = 2048;
constexpr int D_   = 2048;
constexpr int NH_  = 32;
constexpr int NKV_ = 8;
constexpr int HD_  = 64;
constexpr int BS_  = B_ * S_;          // 4096 rows
constexpr int DKV_ = NKV_ * HD_;       // 512

__device__ __forceinline__ unsigned short f32_to_bf16_rne(float f) {
  unsigned u = __float_as_uint(f);
  u += 0x7FFFu + ((u >> 16) & 1u);
  return (unsigned short)(u >> 16);
}

__device__ __forceinline__ v16bf frag_cast(v8u u) {
  return __builtin_bit_cast(v16bf, u);
}

// 16-bit A-matrix (16x32) per-lane K offsets: VGPR j -> K=2j(+16*half),
// contiguous order matching the B-operand fills.
__device__ __forceinline__ int a_koff(int j, int half) {
  return 2 * j + 16 * half;
}

// ---- DPP16 ROW_XMASK cross-lane moves (pure VALU, co-executes with WMMA) ---
template <int CTRL>
__device__ __forceinline__ float dpp_move(float v) {
  return __int_as_float(__builtin_amdgcn_update_dpp(
      0, __float_as_int(v), CTRL, 0xf, 0xf, true));
}
// Reduce across the 16-lane half (masks 1,2,4,8 stay within a DPP row).
__device__ __forceinline__ float half16_max(float v) {
  v = fmaxf(v, dpp_move<0x161>(v));   // ROW_XMASK 1
  v = fmaxf(v, dpp_move<0x162>(v));   // ROW_XMASK 2
  v = fmaxf(v, dpp_move<0x164>(v));   // ROW_XMASK 4
  v = fmaxf(v, dpp_move<0x168>(v));   // ROW_XMASK 8
  return v;
}
__device__ __forceinline__ float half16_sum(float v) {
  v += dpp_move<0x161>(v);
  v += dpp_move<0x162>(v);
  v += dpp_move<0x164>(v);
  v += dpp_move<0x168>(v);
  return v;
}

// ---------------------------------------------------------------------------
// fp32 -> bf16 elementwise convert
// ---------------------------------------------------------------------------
__global__ void f32_to_bf16_k(const float* __restrict__ in,
                              unsigned short* __restrict__ out, int n) {
  int i = blockIdx.x * 256 + threadIdx.x;
  if (i < n) out[i] = f32_to_bf16_rne(in[i]);
}

// ---------------------------------------------------------------------------
// V transpose: vb [B*S, NKV*HD] -> vt [B][NKV][HD][S]  (bf16)
// Makes PV B-operand pairs contiguous along the key axis.
// ---------------------------------------------------------------------------
__global__ void v_transpose_k(const unsigned short* __restrict__ vb,
                              unsigned short* __restrict__ vt) {
  int i = blockIdx.x * 256 + threadIdx.x;          // over BS_*DKV_
  if (i >= BS_ * DKV_) return;
  int row = i >> 9, col = i & 511;                 // DKV_ = 512
  int b = row >> 11, s = row & 2047;               // S_ = 2048
  int n = col >> 6, hd = col & 63;
  vt[((size_t)((b * NKV_ + n) * HD_ + hd) << 11) + s] = vb[i];
}

// ---------------------------------------------------------------------------
// Tiled bf16 GEMM:  C[M,N] = out_scale * (A[M,K] * B[K,N]), fp32 accumulate
// via v_wmma_f32_16x16x32_bf16.  Block = 128 threads (4 waves), tile 64x64,
// K-step 32.  B is transposed into LDS so both operand fragments are
// contiguous-pair ds_load_b128s.
// ---------------------------------------------------------------------------
template <bool OUT_BF16>
__global__ __launch_bounds__(128) void gemm_bf16_wmma(
    const unsigned short* __restrict__ A,   // [M,K] bf16 row-major
    const unsigned short* __restrict__ Bm,  // [K,N] bf16 row-major
    void* __restrict__ Cout, int M, int N, int K, float out_scale) {
  __shared__ unsigned short As[64][36];     // [m][k], +4 pad (72B rows)
  __shared__ unsigned short Bs[64][36];     // [n][k] (transposed), +4 pad

  const int tid  = threadIdx.x;
  const int wave = tid >> 5;
  const int l    = tid & 31;
  const int lo   = l & 15;
  const int hf   = (l >> 4) & 1;
  const int m0   = blockIdx.y * 64;
  const int n0   = blockIdx.x * 64;

  v8f acc[4] = {};

  for (int k0 = 0; k0 < K; k0 += 32) {
    // A tile 64x32 bf16: 1024 dwords, 8 per thread, coalesced along K.
#pragma unroll
    for (int i = 0; i < 8; ++i) {
      int idx = tid + i * 128;
      int r = idx >> 4, cu = idx & 15;
      unsigned v = *(const unsigned*)(&A[(size_t)(m0 + r) * K + k0 + 2 * cu]);
      *(unsigned*)(&As[r][2 * cu]) = v;
    }
    // B tile 32x64 bf16, coalesced along N, stored transposed [n][k].
#pragma unroll
    for (int i = 0; i < 8; ++i) {
      int idx = tid + i * 128;
      int kk = idx >> 5, nu = idx & 31;
      unsigned v = *(const unsigned*)(&Bm[(size_t)(k0 + kk) * N + n0 + 2 * nu]);
      Bs[2 * nu + 0][kk] = (unsigned short)(v & 0xFFFFu);
      Bs[2 * nu + 1][kk] = (unsigned short)(v >> 16);
    }
    if (k0 + 32 < K) {  // global_prefetch_b8 the next K tile
      __builtin_prefetch(&A[(size_t)(m0 + (tid >> 1)) * K + k0 + 32], 0, 1);
      __builtin_prefetch(&Bm[(size_t)(k0 + 32 + (tid & 31)) * N + n0], 0, 1);
    }
    __syncthreads();

    // A fragment: row = 16*wave + lane%16, contiguous K order (K = 16*half + 2j).
    v8u au;
#pragma unroll
    for (int j = 0; j < 8; ++j)
      au[j] = *(const unsigned*)(&As[wave * 16 + lo][a_koff(j, hf)]);
    v16bf av = frag_cast(au);

#pragma unroll
    for (int c = 0; c < 4; ++c) {
      // B fragment (32x16): lanes0-15 K=0..15, lanes16-31 K=16..31 (pairs).
      v8u bu;
#pragma unroll
      for (int j = 0; j < 8; ++j)
        bu[j] = *(const unsigned*)(&Bs[c * 16 + lo][2 * j + 16 * hf]);
      acc[c] = __builtin_amdgcn_wmma_f32_16x16x32_bf16(
          false, av, false, frag_cast(bu), (short)0, acc[c], false, false);
    }
    __syncthreads();
  }

  // C layout: VGPR r -> row r (+8 for lanes 16-31), col = lane%16.
#pragma unroll
  for (int c = 0; c < 4; ++c) {
#pragma unroll
    for (int r = 0; r < 8; ++r) {
      size_t row = (size_t)(m0 + wave * 16 + r + 8 * hf);
      int col = n0 + c * 16 + lo;
      float v = acc[c][r] * out_scale;
      if (OUT_BF16)
        ((unsigned short*)Cout)[row * N + col] = f32_to_bf16_rne(v);
      else
        ((float*)Cout)[row * N + col] = v;
    }
  }
}

// ---------------------------------------------------------------------------
// Fused flash attention (non-causal).  One wave owns 16 query rows x one
// head: acc = 16x64 fp32 (4 C frags).  Per 32-key step: 2 score tiles
// (2 WMMAs each, Q*K^T; 1/sqrt(HD) pre-folded into Q), online softmax in
// fp32 with DPP row_xmask reductions (VALU, overlaps WMMA), P re-laid out
// C->A via a per-wave LDS tile, then 4 PV WMMAs against V^T.  Scores never
// hit HBM.
// ---------------------------------------------------------------------------
__global__ __launch_bounds__(128) void flash_gqa(
    const unsigned short* __restrict__ qb,  // [BS, NH*HD] bf16 (pre-scaled)
    const unsigned short* __restrict__ kb,  // [BS, NKV*HD] bf16
    const unsigned short* __restrict__ vt,  // [B][NKV][HD][S] bf16
    unsigned short* __restrict__ ao) {      // [BS, NH*HD] bf16
  __shared__ unsigned short Pl[4][16][36];  // per-wave P tile (16x32 + pad)

  const int wave = threadIdx.x >> 5;
  const int l    = threadIdx.x & 31;
  const int lo   = l & 15;
  const int hf   = (l >> 4) & 1;

  const int flat = blockIdx.x * 4 + wave;   // (b, h, qblk)
  const int qblk = flat & 127;              // S/16 = 128
  const int h    = (flat >> 7) & 31;
  const int b    = flat >> 12;
  const int n    = h >> 2;                  // kv head, G = 4

  // Q fragments (depth 0-31 and 32-63), loaded once.
  const size_t qrow = (size_t)(b * S_ + qblk * 16 + lo);
  v8u qulo, quhi;
#pragma unroll
  for (int j = 0; j < 8; ++j) {
    int ko = a_koff(j, hf);
    qulo[j] = *(const unsigned*)(&qb[qrow * (size_t)D_ + h * HD_ + ko]);
    quhi[j] = *(const unsigned*)(&qb[qrow * (size_t)D_ + h * HD_ + 32 + ko]);
  }
  const v16bf qlo = frag_cast(qulo), qhi = frag_cast(quhi);

  v8f acc[4] = {};
  float mrun[8], lrun[8];
#pragma unroll
  for (int r = 0; r < 8; ++r) { mrun[r] = -3.0e38f; lrun[r] = 0.0f; }

  const unsigned short* kbase = kb + (size_t)b * S_ * DKV_ + n * HD_;
  const unsigned short* vbase = vt + (size_t)(b * NKV_ + n) * HD_ * S_;

  for (int key0 = 0; key0 < S_; key0 += 32) {
    // ---- scores: two 16-key tiles, S = (Q/sqrt(HD)) * K^T ----
    v8f s[2];
#pragma unroll
    for (int t = 0; t < 2; ++t) {
      int key = key0 + t * 16 + lo;        // B-operand: lane = key column
      v8u klu, khu;
#pragma unroll
      for (int j = 0; j < 8; ++j) {
        int ko = 2 * j + 16 * hf;          // depth pairs, contiguous in K row
        klu[j] = *(const unsigned*)(&kbase[(size_t)key * DKV_ + ko]);
        khu[j] = *(const unsigned*)(&kbase[(size_t)key * DKV_ + 32 + ko]);
      }
      v8f z = {};
      z = __builtin_amdgcn_wmma_f32_16x16x32_bf16(
          false, qlo, false, frag_cast(klu), (short)0, z, false, false);
      z = __builtin_amdgcn_wmma_f32_16x16x32_bf16(
          false, qhi, false, frag_cast(khu), (short)0, z, false, false);
      s[t] = z;
    }

    // ---- online softmax (rows r+8*hf, columns across 16-lane half) ----
#pragma unroll
    for (int r = 0; r < 8; ++r) {
      float mx = half16_max(fmaxf(s[0][r], s[1][r]));
      float mnew  = fmaxf(mrun[r], mx);
      float alpha = __expf(mrun[r] - mnew);
      mrun[r] = mnew;
      float p0 = __expf(s[0][r] - mnew);
      float p1 = __expf(s[1][r] - mnew);
      float rs = half16_sum(p0 + p1);
      lrun[r] = lrun[r] * alpha + rs;
#pragma unroll
      for (int c = 0; c < 4; ++c) acc[c][r] *= alpha;
      int prow = r + 8 * hf;               // C layout -> LDS [row][key]
      Pl[wave][prow][lo]      = f32_to_bf16_rne(p0);
      Pl[wave][prow][16 + lo] = f32_to_bf16_rne(p1);
    }

    // ---- re-read P in A layout (same-wave LDS, dscnt-ordered) ----
    v8u pu;
#pragma unroll
    for (int j = 0; j < 8; ++j)
      pu[j] = *(const unsigned*)(&Pl[wave][lo][a_koff(j, hf)]);
    const v16bf pv = frag_cast(pu);

    // ---- acc += P(16x32) * V(32x64), B-operand from V^T ----
#pragma unroll
    for (int c = 0; c < 4; ++c) {
      v8u vu;
#pragma unroll
      for (int j = 0; j < 8; ++j)
        vu[j] = *(const unsigned*)(
            &vbase[(size_t)(c * 16 + lo) * S_ + key0 + 2 * j + 16 * hf]);
      acc[c] = __builtin_amdgcn_wmma_f32_16x16x32_bf16(
          false, pv, false, frag_cast(vu), (short)0, acc[c], false, false);
    }
  }

  // ---- normalize and store bf16 attention output ----
#pragma unroll
  for (int r = 0; r < 8; ++r) {
    float inv = 1.0f / lrun[r];
    size_t row = (size_t)(b * S_ + qblk * 16 + r + 8 * hf);
#pragma unroll
    for (int c = 0; c < 4; ++c) {
      int col = h * HD_ + c * 16 + lo;
      ao[row * (size_t)D_ + col] = f32_to_bf16_rne(acc[c][r] * inv);
    }
  }
}

// ---------------------------------------------------------------------------
// Launch
// ---------------------------------------------------------------------------
extern "C" void kernel_launch(void* const* d_in, const int* in_sizes, int n_in,
                              void* d_out, int out_size, void* d_ws,
                              size_t ws_size, hipStream_t stream) {
  (void)in_sizes; (void)n_in; (void)out_size; (void)ws_size;
  const float* x  = (const float*)d_in[0];
  const float* Wq = (const float*)d_in[1];
  const float* Wk = (const float*)d_in[2];
  const float* Wv = (const float*)d_in[3];
  const float* Wo = (const float*)d_in[4];
  float* out = (float*)d_out;

  unsigned char* w = (unsigned char*)d_ws;
  auto alloc = [&](size_t elems) {
    unsigned short* p = (unsigned short*)w;
    w += elems * sizeof(unsigned short);
    return p;
  };
  unsigned short* xb   = alloc((size_t)BS_ * D_);
  unsigned short* Wqb  = alloc((size_t)D_ * D_);
  unsigned short* Wkb  = alloc((size_t)D_ * DKV_);
  unsigned short* Wvb  = alloc((size_t)D_ * DKV_);
  unsigned short* Wob  = alloc((size_t)D_ * D_);
  unsigned short* qbuf = alloc((size_t)BS_ * D_);
  unsigned short* kbuf = alloc((size_t)BS_ * DKV_);
  unsigned short* vbuf = alloc((size_t)BS_ * DKV_);
  unsigned short* vtb  = alloc((size_t)BS_ * DKV_);
  unsigned short* aob  = alloc((size_t)BS_ * D_);

  auto cvt = [&](const float* src, unsigned short* dst, int n) {
    f32_to_bf16_k<<<(n + 255) / 256, 256, 0, stream>>>(src, dst, n);
  };
  cvt(x,  xb,  BS_ * D_);
  cvt(Wq, Wqb, D_ * D_);
  cvt(Wk, Wkb, D_ * DKV_);
  cvt(Wv, Wvb, D_ * DKV_);
  cvt(Wo, Wob, D_ * D_);

  // Projections (bf16 out, fp32 accumulate).  1/sqrt(HD) folded into Q.
  gemm_bf16_wmma<true><<<dim3(D_ / 64, BS_ / 64), 128, 0, stream>>>(
      xb, Wqb, qbuf, BS_, D_, D_, 0.125f);
  gemm_bf16_wmma<true><<<dim3(DKV_ / 64, BS_ / 64), 128, 0, stream>>>(
      xb, Wkb, kbuf, BS_, DKV_, D_, 1.0f);
  gemm_bf16_wmma<true><<<dim3(DKV_ / 64, BS_ / 64), 128, 0, stream>>>(
      xb, Wvb, vbuf, BS_, DKV_, D_, 1.0f);

  v_transpose_k<<<(BS_ * DKV_ + 255) / 256, 256, 0, stream>>>(vbuf, vtb);

  // Fused attention: B*NH*(S/16) = 8192 waves, 4 waves per block.
  flash_gqa<<<(B_ * NH_ * (S_ / 16)) / 4, 128, 0, stream>>>(qbuf, kbuf, vtb,
                                                            aob);

  // Output projection, fp32 result straight to d_out.
  gemm_bf16_wmma<false><<<dim3(D_ / 64, BS_ / 64), 128, 0, stream>>>(
      aob, Wob, out, BS_, D_, D_, 1.0f);
}
